// W4A8BF32OF32Linear_81913616270134
// MI455X (gfx1250) — 8-bit matrix operand order repaired and checked by reading, compiled, not run on hardware
//
#include <hip/hip_runtime.h>

typedef int v8i __attribute__((ext_vector_type(8)));

#define TILE_M 128
#define TILE_N 128
#define TILE_K 64
#define LDS_STRIDE 80
#define BUF_BYTES (128 * LDS_STRIDE)

union FragA { v8i v; unsigned long long u64[4]; };
union FragB { v8i v; uint4 u128[2]; };

__global__ void pack_x_kernel(const int* __restrict__ x, char* __restrict__ x8, int total4) {
    int idx = blockIdx.x * blockDim.x + threadIdx.x;
    if (idx >= total4) return;
    const int4 v = ((const int4*)x)[idx];
    int packed = (v.x & 0xFF) | ((v.y & 0xFF) << 8) | ((v.z & 0xFF) << 16) | ((v.w & 0xFF) << 24);
    ((int*)x8)[idx] = packed;
}

__global__ void dequant_w_kernel(const int* __restrict__ wp,
                                 const int* __restrict__ scales8,
                                 const int* __restrict__ zeros,
                                 char* __restrict__ w8,
                                 int K, int G, int gs, int total4) {
    int idx = blockIdx.x * blockDim.x + threadIdx.x;
    if (idx >= total4) return;
    int kflat = idx * 4;
    int o = kflat / K;
    int k = kflat - o * K;
    int g = k / gs;
    int z = zeros[o * G + g];
    int s = scales8[o * G + g];
    int p0 = wp[idx * 2];
    int p1 = wp[idx * 2 + 1];
    int q0 = ((p0 & 0xF) - z) * s;
    int q1 = (((p0 >> 4) & 0xF) - z) * s;
    int q2 = ((p1 & 0xF) - z) * s;
    int q3 = (((p1 >> 4) & 0xF) - z) * s;
    int packed = (q0 & 0xFF) | ((q1 & 0xFF) << 8) | ((q2 & 0xFF) << 16) | ((q3 & 0xFF) << 24);
    ((int*)w8)[idx] = packed;
}

__global__ __launch_bounds__(256)
void w4a8_gemm_kernel(const char* __restrict__ x8, const char* __restrict__ w8,
                      const float* __restrict__ avec, const float* __restrict__ bvec,
                      const float* __restrict__ bias, float* __restrict__ out,
                      int K, int O) {
    __shared__ unsigned char lA[2 * BUF_BYTES];
    __shared__ unsigned char lB[2 * BUF_BYTES];

    const int tid  = threadIdx.x;
    const int lane = tid & 31;
    const int wave = tid >> 5;
    const int wm   = wave >> 1;
    const int wn   = wave & 1;

    const int m0 = blockIdx.y * TILE_M;
    const int o0 = blockIdx.x * TILE_N;

    const int lhalf = lane >> 4;
    const int lmod  = lane & 15;

    v8i acc[2][4];
#pragma unroll
    for (int s = 0; s < 2; ++s)
#pragma unroll
        for (int t = 0; t < 4; ++t) {
            v8i z = {0, 0, 0, 0, 0, 0, 0, 0};
            acc[s][t] = z;
        }

    const int c0   = tid * 2;
    const int row0 = c0 >> 2;
    const int ko0  = (c0 & 3) * 16;
    const int row1 = (c0 + 1) >> 2;
    const int ko1  = ((c0 + 1) & 3) * 16;

    const char* gA0 = x8 + (size_t)(m0 + row0) * K + ko0;
    const char* gA1 = x8 + (size_t)(m0 + row1) * K + ko1;
    const char* gB0 = w8 + (size_t)(o0 + row0) * K + ko0;
    const char* gB1 = w8 + (size_t)(o0 + row1) * K + ko1;

    const int stA0 = row0 * LDS_STRIDE + ko0;
    const int stA1 = row1 * LDS_STRIDE + ko1;

    const int fA0 = (wm * 32 + lmod) * LDS_STRIDE + lhalf * 16;
    const int fB0 = (wn * 64 + lmod) * LDS_STRIDE + lhalf * 16;

    const int kSteps = K / TILE_K;

    {
        uint4 va0 = *(const uint4*)(gA0);
        uint4 va1 = *(const uint4*)(gA1);
        uint4 vb0 = *(const uint4*)(gB0);
        uint4 vb1 = *(const uint4*)(gB1);
        *(uint4*)(lA + stA0) = va0;
        *(uint4*)(lA + stA1) = va1;
        *(uint4*)(lB + stA0) = vb0;
        *(uint4*)(lB + stA1) = vb1;
    }
    __syncthreads();

    for (int kt = 0; kt < kSteps; ++kt) {
        const int cur = kt & 1;
        const int curOff = cur * BUF_BYTES;
        const int nxtOff = (cur ^ 1) * BUF_BYTES;
        const bool have_next = (kt + 1 < kSteps);

        uint4 va0, va1, vb0, vb1;
        if (have_next) {
            const int kn = (kt + 1) * TILE_K;
            va0 = *(const uint4*)(gA0 + kn);
            va1 = *(const uint4*)(gA1 + kn);
            vb0 = *(const uint4*)(gB0 + kn);
            vb1 = *(const uint4*)(gB1 + kn);
            if (kt + 2 < kSteps) {
                __builtin_prefetch(gA0 + kn + TILE_K, 0, 3);
                __builtin_prefetch(gB0 + kn + TILE_K, 0, 3);
            }
        }

        FragA af[2];
#pragma unroll
        for (int s = 0; s < 2; ++s) {
            const unsigned long long* p =
                (const unsigned long long*)(lA + curOff + fA0 + s * 16 * LDS_STRIDE);
            af[s].u64[0] = p[0];
            af[s].u64[1] = p[1];
            af[s].u64[2] = p[4];
            af[s].u64[3] = p[5];
        }
        FragB bf[4];
#pragma unroll
        for (int t = 0; t < 4; ++t) {
            const uint4* p = (const uint4*)(lB + curOff + fB0 + t * 16 * LDS_STRIDE);
            bf[t].u128[0] = p[0];
            bf[t].u128[1] = p[2];
        }

#pragma unroll
        for (int s = 0; s < 2; ++s)
#pragma unroll
            for (int t = 0; t < 4; ++t)
                acc[s][t] = __builtin_amdgcn_wmma_i32_16x16x64_iu8(
                     true, af[s].v,  true, bf[t].v,
                    acc[s][t],  false,  false);

        if (have_next) {
            *(uint4*)(lA + nxtOff + stA0) = va0;
            *(uint4*)(lA + nxtOff + stA1) = va1;
            *(uint4*)(lB + nxtOff + stA0) = vb0;
            *(uint4*)(lB + nxtOff + stA1) = vb1;
        }
        __syncthreads();
    }

#pragma unroll
    for (int t = 0; t < 4; ++t) {
        int col = o0 + wn * 64 + t * 16 + lmod;
        float sc = avec[col];
        float bs = bvec[col] + bias[col];
#pragma unroll
        for (int s = 0; s < 2; ++s) {
            int rbase = m0 + wm * 32 + s * 16 + (lhalf ? 8 : 0);
#pragma unroll
            for (int r = 0; r < 8; ++r) {
                float v = (float)acc[s][t][r] * sc + bs;
                out[(size_t)(rbase + r) * O + col] = v;
            }
        }
    }
}

extern "C" void kernel_launch(void* const* d_in, const int* in_sizes, int n_in,
                              void* d_out, int out_size, void* d_ws, size_t ws_size,
                              hipStream_t stream) {
    const int*   x       = (const int*)d_in[0];
    const int*   wp      = (const int*)d_in[1];
    const float* bias    = (const float*)d_in[2];
    const float* avec    = (const float*)d_in[3];
    const float* bvec    = (const float*)d_in[4];
    const int*   scales8 = (const int*)d_in[5];
    const int*   zeros   = (const int*)d_in[6];

    const int O  = in_sizes[2];
    const int K  = (int)(2LL * in_sizes[1] / O);
    const int N  = (int)((long long)in_sizes[0] / K);
    const int G  = in_sizes[5] / O;
    const int gs = K / G;

    char* x8 = (char*)d_ws;
    char* w8 = (char*)d_ws + (size_t)N * K;

    const int totalX4 = (N * K) / 4;
    pack_x_kernel<<<(totalX4 + 255) / 256, 256, 0, stream>>>(x, x8, totalX4);

    const int totalW4 = (O * K) / 4;
    dequant_w_kernel<<<(totalW4 + 255) / 256, 256, 0, stream>>>(
        wp, scales8, zeros, w8, K, G, gs, totalW4);

    dim3 grid(O / TILE_N, N / TILE_M);
    w4a8_gemm_kernel<<<grid, 256, 0, stream>>>(x8, w8, avec, bvec, bias,
                                               (float*)d_out, K, O);
}
